// Decoder_8478265442540
// MI455X (gfx1250) — matrix operand order repaired and checked by reading, not compiled, not run on hardware
//
#include <hip/hip_runtime.h>
#include <hip/hip_bf16.h>

typedef __bf16 v16bf __attribute__((ext_vector_type(16)));
typedef __bf16 bf8v  __attribute__((ext_vector_type(8)));
typedef float  v8f   __attribute__((ext_vector_type(8)));

#define T_  32
#define B_  64
#define S_  64
#define E_  512
#define H_  1024
#define G4H 4096   // 4*H

// ---------------------------------------------------------------------------
// fp32 -> bf16 convert
__global__ void k_cvt(const float* __restrict__ s, __bf16* __restrict__ d, int n) {
    int i = blockIdx.x * blockDim.x + threadIdx.x;
    if (i < n) d[i] = (__bf16)s[i];
}

// pack [W_ih | W_hh] along K into bf16, row-major (N=4096, K=Kih+Khh)
__global__ void k_pack_w(const float* __restrict__ wih, const float* __restrict__ whh,
                         __bf16* __restrict__ dst, int Kih, int Khh) {
    int K = Kih + Khh;
    int i = blockIdx.x * blockDim.x + threadIdx.x;
    if (i >= G4H * K) return;
    int k = i % K, row = i / K;
    float v = (k < Kih) ? wih[(size_t)row * Kih + k]
                        : whh[(size_t)row * Khh + (k - Kih)];
    dst[i] = (__bf16)v;
}

// build x0 = [ emb[input[t,b]] (512) | out_feed (1024) | h0_prev (1024) ] as bf16
__global__ void k_build_x0(const int* __restrict__ input, const float* __restrict__ emb,
                           const float* __restrict__ outf32, const float* __restrict__ hbuf,
                           __bf16* __restrict__ x0, int t) {
    int i = blockIdx.x * blockDim.x + threadIdx.x;
    if (i >= B_ * 2560) return;
    int b = i / 2560, k = i % 2560;
    float v;
    if (k < E_) {
        int idx = input[t * B_ + b];
        v = emb[(size_t)idx * E_ + k];
    } else if (k < E_ + H_) {
        v = outf32[b * H_ + (k - E_)];
    } else {
        v = hbuf[b * H_ + (k - E_ - H_)];   // layer-0 h_prev
    }
    x0[i] = (__bf16)v;
}

// ---------------------------------------------------------------------------
// WMMA bf16 GEMM, M = 64 fixed:  C(64,N) = A(64,K) * B(N,K)^T
// One wave computes the full 64x16 output strip for one N tile: the B fragment
// is loaded once per k-step and reused by 4 WMMAs (4 independent acc chains).
// Software-pipelined: iteration k+1's fragments are loaded before iteration
// k's WMMAs so the per-WMMA wait covers loads issued a full iteration earlier.
struct Frags { v16bf b, a0, a1, a2, a3; };

template <int K>
__device__ __forceinline__ Frags load_frags(const __bf16* __restrict__ Arow,
                                            const __bf16* __restrict__ Brow,
                                            int k0, int half) {
    Frags f;
    bf8v b0 = *(const bf8v*)(Brow + k0 + half * 8);
    bf8v b1 = *(const bf8v*)(Brow + k0 + 16 + half * 8);
#pragma unroll
    for (int j = 0; j < 8; ++j) { f.b[j] = b0[j]; f.b[j + 8] = b1[j]; }
#define MK_AFRAG(av, MOFF)                                                        \
    {   bf8v a0 = *(const bf8v*)(Arow + (MOFF) + k0 + half * 8);                  \
        bf8v a1 = *(const bf8v*)(Arow + (MOFF) + k0 + 16 + half * 8);             \
        _Pragma("unroll")                                                         \
        for (int j = 0; j < 8; ++j) { av[j] = a0[j]; av[j + 8] = a1[j]; } }
    MK_AFRAG(f.a0, 0 * 16 * K) MK_AFRAG(f.a1, 1 * 16 * K)
    MK_AFRAG(f.a2, 2 * 16 * K) MK_AFRAG(f.a3, 3 * 16 * K)
#undef MK_AFRAG
    return f;
}

template <int N, int K>
__global__ void k_gemm_bf16_m64(const __bf16* __restrict__ A, const __bf16* __restrict__ Bw,
                                float* __restrict__ C) {
    int gtid = blockIdx.x * blockDim.x + threadIdx.x;
    int wave = gtid >> 5;           // one wave per N tile
    int lane = gtid & 31;
    if (wave >= (N >> 4)) return;   // grids are exact; never diverges mid-wave
    int half = lane >> 4;
    int r    = lane & 15;

    const __bf16* Arow = A  + (size_t)r * K;                 // m-tile 0 row; tiles 1..3 at +16*K
    const __bf16* Brow = Bw + (size_t)(wave * 16 + r) * K;

    v8f acc0 = {}, acc1 = {}, acc2 = {}, acc3 = {};
    Frags cur = load_frags<K>(Arow, Brow, 0, half);
#pragma unroll 2
    for (int k0 = 0; k0 < K; k0 += 32) {
        // prefetch weight stream ahead (speculative: dropped if OOB)
        __builtin_prefetch(Brow + k0 + 64, 0, 1);
        // issue next iteration's loads before this iteration's WMMAs
        int kn = (k0 + 32 < K) ? (k0 + 32) : k0;
        Frags nxt = load_frags<K>(Arow, Brow, kn, half);

        acc0 = __builtin_amdgcn_wmma_f32_16x16x32_bf16(false, cur.a0, false, cur.b, (short)0, acc0, false, false);
        acc1 = __builtin_amdgcn_wmma_f32_16x16x32_bf16(false, cur.a1, false, cur.b, (short)0, acc1, false, false);
        acc2 = __builtin_amdgcn_wmma_f32_16x16x32_bf16(false, cur.a2, false, cur.b, (short)0, acc2, false, false);
        acc3 = __builtin_amdgcn_wmma_f32_16x16x32_bf16(false, cur.a3, false, cur.b, (short)0, acc3, false, false);
        cur = nxt;
    }
    // D layout: VGPR v -> row m = mTile*16 + v + 8*half, column n = r
    float* Cc = C + (size_t)wave * 16 + r;
#pragma unroll
    for (int v = 0; v < 8; ++v) {
        Cc[(size_t)(0  + v + 8 * half) * N] = acc0[v];
        Cc[(size_t)(16 + v + 8 * half) * N] = acc1[v];
        Cc[(size_t)(32 + v + 8 * half) * N] = acc2[v];
        Cc[(size_t)(48 + v + 8 * half) * N] = acc3[v];
    }
}

// ---------------------------------------------------------------------------
__device__ __forceinline__ float sigf(float x) { return 1.0f / (1.0f + __expf(-x)); }

// LSTM layer 0 cell; also stages x1 = [h0_new | h1_prev] bf16 for layer-1 GEMM
__global__ void k_lstm0(const float* __restrict__ gates, const float* __restrict__ bias,
                        float* __restrict__ hbuf, float* __restrict__ cbuf,
                        __bf16* __restrict__ x1) {
    int i = blockIdx.x * blockDim.x + threadIdx.x;
    if (i >= B_ * H_) return;
    int b = i >> 10, j = i & 1023;
    const float* g = gates + (size_t)b * G4H;
    float gi = g[j]            + bias[j];
    float gf = g[H_ + j]       + bias[H_ + j];
    float gg = g[2 * H_ + j]   + bias[2 * H_ + j];
    float go = g[3 * H_ + j]   + bias[3 * H_ + j];
    float cn = sigf(gf) * cbuf[i] + sigf(gi) * tanhf(gg);
    float hn = sigf(go) * tanhf(cn);
    cbuf[i] = cn;
    hbuf[i] = hn;
    x1[b * 2048 + j]        = (__bf16)hn;
    x1[b * 2048 + H_ + j]   = (__bf16)hbuf[B_ * H_ + i];   // layer-1 h_prev
}

// LSTM layer 1 cell; stages h1 bf16 and the query half of the attn concat
__global__ void k_lstm1(const float* __restrict__ gates, const float* __restrict__ bias,
                        float* __restrict__ hbufL, float* __restrict__ cbufL,
                        __bf16* __restrict__ h1bf, __bf16* __restrict__ xat) {
    int i = blockIdx.x * blockDim.x + threadIdx.x;
    if (i >= B_ * H_) return;
    int b = i >> 10, j = i & 1023;
    const float* g = gates + (size_t)b * G4H;
    float gi = g[j]            + bias[j];
    float gf = g[H_ + j]       + bias[H_ + j];
    float gg = g[2 * H_ + j]   + bias[2 * H_ + j];
    float go = g[3 * H_ + j]   + bias[3 * H_ + j];
    float cn = sigf(gf) * cbufL[i] + sigf(gi) * tanhf(gg);
    float hn = sigf(go) * tanhf(cn);
    cbufL[i] = cn;
    hbufL[i] = hn;
    h1bf[i]  = (__bf16)hn;
    xat[b * 2048 + H_ + j] = (__bf16)hn;    // concat([wctx, query]) query part
}

// ---------------------------------------------------------------------------
// fused scores -> softmax -> (optional) weighted context. One block per batch b.
// q:(B,H) fp32, ctx:(S,B,H) fp32, attn_dst:(B,S), wctx_dst: bf16 into xat[b,0:H]
__global__ void k_attn(const float* __restrict__ q, const float* __restrict__ ctx,
                       float* __restrict__ attn_dst, __bf16* __restrict__ wctx_dst) {
    __shared__ float sc[S_];
    __shared__ float s_inv;
    int b = blockIdx.x;
    int tid = threadIdx.x;
    int wave = tid >> 5, lane = tid & 31;

    // 8 waves x 8 scores each
    for (int r = 0; r < 8; ++r) {
        int s = wave * 8 + r;
        const float* c = ctx + ((size_t)s * B_ + b) * H_;
        const float* qq = q + (size_t)b * H_;
        float sum = 0.f;
        for (int d = lane; d < H_; d += 32) sum += qq[d] * c[d];
#pragma unroll
        for (int off = 16; off; off >>= 1) sum += __shfl_xor(sum, off, 32);
        if (lane == 0) sc[s] = sum;
    }
    __syncthreads();
    if (tid == 0) {
        float m = sc[0];
        for (int s = 1; s < S_; ++s) m = fmaxf(m, sc[s]);
        float tot = 0.f;
        for (int s = 0; s < S_; ++s) { float e = __expf(sc[s] - m); sc[s] = e; tot += e; }
        s_inv = 1.0f / tot;
    }
    __syncthreads();
    if (tid < S_) {
        float a = sc[tid] * s_inv;
        sc[tid] = a;
        attn_dst[b * S_ + tid] = a;
    }
    __syncthreads();
    if (wctx_dst) {
        for (int d = tid; d < H_; d += 256) {
            float acc = 0.f;
            for (int s = 0; s < S_; ++s)
                acc += sc[s] * ctx[((size_t)s * B_ + b) * H_ + d];
            wctx_dst[b * 2048 + d] = (__bf16)acc;
        }
    }
}

// out = tanh(outlin); writes outputs[t] (d_out), out_feed fp32, out bf16
__global__ void k_tanh_out(const float* __restrict__ outlin, float* __restrict__ out_dst,
                           float* __restrict__ outf32, __bf16* __restrict__ outbf) {
    int i = blockIdx.x * blockDim.x + threadIdx.x;
    if (i >= B_ * H_) return;
    float v = tanhf(outlin[i]);
    out_dst[i] = v;
    outf32[i]  = v;
    outbf[i]   = (__bf16)v;
}

// ---------------------------------------------------------------------------
extern "C" void kernel_launch(void* const* d_in, const int* in_sizes, int n_in,
                              void* d_out, int out_size, void* d_ws, size_t ws_size,
                              hipStream_t stream) {
    const int*   input     = (const int*)d_in[0];
    const float* h0        = (const float*)d_in[1];
    const float* c0        = (const float*)d_in[2];
    const float* ctx       = (const float*)d_in[3];
    const float* init_out  = (const float*)d_in[4];
    const float* emb       = (const float*)d_in[5];
    const float* Wih0      = (const float*)d_in[6];
    const float* Whh0      = (const float*)d_in[7];
    const float* b0        = (const float*)d_in[8];
    const float* Wih1      = (const float*)d_in[9];
    const float* Whh1      = (const float*)d_in[10];
    const float* b1        = (const float*)d_in[11];
    const float* attn_in_w = (const float*)d_in[12];
    const float* attn_out_w= (const float*)d_in[13];
    const float* copy_in_w = (const float*)d_in[14];
    float* out = (float*)d_out;

    // d_out layout (floats): outputs | h_T | c_T | attn_std | attn_copy
    const size_t outputsOff = 0;
    const size_t hOff       = (size_t)T_ * B_ * H_;
    const size_t cOff       = hOff + 2ull * B_ * H_;
    const size_t stdOff     = cOff + 2ull * B_ * H_;
    const size_t copyOff    = stdOff + (size_t)T_ * B_ * S_;

    // workspace carve-up
    char* p = (char*)d_ws;
    auto carve = [&](size_t bytes) -> void* {
        void* r = (void*)p;
        p += (bytes + 255) & ~(size_t)255;
        return r;
    };
    __bf16* w0    = (__bf16*)carve((size_t)G4H * 2560 * 2);  // [W_ih0|W_hh0]
    __bf16* w1    = (__bf16*)carve((size_t)G4H * 2048 * 2);  // [W_ih1|W_hh1]
    __bf16* wai   = (__bf16*)carve((size_t)H_ * H_ * 2);
    __bf16* wao   = (__bf16*)carve((size_t)H_ * 2 * H_ * 2);
    __bf16* wci   = (__bf16*)carve((size_t)H_ * H_ * 2);
    __bf16* x0    = (__bf16*)carve((size_t)B_ * 2560 * 2);
    __bf16* x1    = (__bf16*)carve((size_t)B_ * 2048 * 2);
    __bf16* xat   = (__bf16*)carve((size_t)B_ * 2048 * 2);
    __bf16* h1bf  = (__bf16*)carve((size_t)B_ * H_ * 2);
    __bf16* outbf = (__bf16*)carve((size_t)B_ * H_ * 2);
    float*  gates = (float*)carve((size_t)B_ * G4H * 4);
    float*  qbuf  = (float*)carve((size_t)B_ * H_ * 4);
    float*  qcbuf = (float*)carve((size_t)B_ * H_ * 4);
    float*  outlin= (float*)carve((size_t)B_ * H_ * 4);
    float*  outf32= (float*)carve((size_t)B_ * H_ * 4);
    float*  hbuf  = (float*)carve(2ull * B_ * H_ * 4);
    float*  cbuf  = (float*)carve(2ull * B_ * H_ * 4);

    const int BT = 256;
    auto grid = [](size_t n) { return (unsigned)((n + 255) / 256); };

    // ---- one-time weight pack / convert (deterministic every call) ----
    k_pack_w<<<grid((size_t)G4H * 2560), BT, 0, stream>>>(Wih0, Whh0, w0, E_ + H_, H_);
    k_pack_w<<<grid((size_t)G4H * 2048), BT, 0, stream>>>(Wih1, Whh1, w1, H_, H_);
    k_cvt<<<grid((size_t)H_ * H_),     BT, 0, stream>>>(attn_in_w,  wai, H_ * H_);
    k_cvt<<<grid((size_t)H_ * 2 * H_), BT, 0, stream>>>(attn_out_w, wao, H_ * 2 * H_);
    k_cvt<<<grid((size_t)H_ * H_),     BT, 0, stream>>>(copy_in_w,  wci, H_ * H_);

    // ---- init recurrent state ----
    hipMemcpyAsync(hbuf, h0, 2ull * B_ * H_ * 4, hipMemcpyDeviceToDevice, stream);
    hipMemcpyAsync(cbuf, c0, 2ull * B_ * H_ * 4, hipMemcpyDeviceToDevice, stream);
    hipMemcpyAsync(outf32, init_out, (size_t)B_ * H_ * 4, hipMemcpyDeviceToDevice, stream);

    // GEMM grids: one wave per N tile (wave handles full M=64 strip)
    const unsigned gGemmBig = (G4H / 16) * 32 / 256;   // 256 waves -> 32 blocks
    const unsigned gGemmSm  = (H_  / 16) * 32 / 256;   // 64 waves  -> 8 blocks
    const unsigned gBH      = grid((size_t)B_ * H_);

    for (int t = 0; t < T_; ++t) {
        // layer 0
        k_build_x0<<<grid((size_t)B_ * 2560), BT, 0, stream>>>(input, emb, outf32, hbuf, x0, t);
        k_gemm_bf16_m64<G4H, 2560><<<gGemmBig, BT, 0, stream>>>(x0, w0, gates);
        k_lstm0<<<gBH, BT, 0, stream>>>(gates, b0, hbuf, cbuf, x1);
        // layer 1
        k_gemm_bf16_m64<G4H, 2048><<<gGemmBig, BT, 0, stream>>>(x1, w1, gates);
        k_lstm1<<<gBH, BT, 0, stream>>>(gates, b1, hbuf + B_ * H_, cbuf + B_ * H_, h1bf, xat);
        // global attention
        k_gemm_bf16_m64<H_, H_><<<gGemmSm, BT, 0, stream>>>(h1bf, wai, qbuf);
        k_attn<<<B_, BT, 0, stream>>>(qbuf, ctx, out + stdOff + (size_t)t * B_ * S_, xat);
        k_gemm_bf16_m64<H_, 2 * H_><<<gGemmSm, BT, 0, stream>>>(xat, wao, outlin);
        k_tanh_out<<<gBH, BT, 0, stream>>>(outlin, out + outputsOff + (size_t)t * B_ * H_,
                                           outf32, outbf);
        // copy attention (weights only)
        k_gemm_bf16_m64<H_, H_><<<gGemmSm, BT, 0, stream>>>(outbf, wci, qcbuf);
        k_attn<<<B_, BT, 0, stream>>>(qcbuf, ctx, out + copyOff + (size_t)t * B_ * S_, nullptr);
    }

    // final states
    hipMemcpyAsync(out + hOff, hbuf, 2ull * B_ * H_ * 4, hipMemcpyDeviceToDevice, stream);
    hipMemcpyAsync(out + cOff, cbuf, 2ull * B_ * H_ * 4, hipMemcpyDeviceToDevice, stream);
}
